// MultiHeadLatentAttention_73435350827095
// MI455X (gfx1250) — matrix operand order repaired and checked by reading, not compiled, not run on hardware
//
#include <hip/hip_runtime.h>
#include <math.h>

#define N_HEAD   16
#define KV_HEADS 4
#define N_EMBED  2048
#define HEAD_DIM 128
#define Q_RANK   256
#define KV_RANK  128
#define B_SZ     8
#define S_LEN    1024
#define BS_TOK   (B_SZ * S_LEN)   // 8192 tokens

typedef __attribute__((ext_vector_type(16))) _Float16 v16h;
typedef __attribute__((ext_vector_type(8)))  _Float16 v8h;
typedef __attribute__((ext_vector_type(8)))  float    v8f;

// ---------------------------------------------------------------------------
// CDNA5 async copy: global -> LDS, 16B per active lane (ASYNCcnt-tracked).
// lds_off is the LDS byte address for this lane; src the global address.
// ---------------------------------------------------------------------------
__device__ __forceinline__ void async_cp_b128(unsigned lds_off, const void* src) {
  asm volatile("global_load_async_to_lds_b128 %0, %1, off"
               :: "v"(lds_off), "v"(src)
               : "memory");
}
__device__ __forceinline__ void wait_asynccnt0() {
  asm volatile("s_wait_asynccnt 0x0" ::: "memory");
}

// ---------------------------------------------------------------------------
// Elementwise f32 -> f16 convert
// ---------------------------------------------------------------------------
__global__ void cvt_f32_f16(const float* __restrict__ in, _Float16* __restrict__ out, int n) {
  int i = blockIdx.x * blockDim.x + threadIdx.x;
  if (i < n) out[i] = (_Float16)in[i];
}

// Transpose-convert: in[K][N] f32 (row-major) -> out[N][K] f16
__global__ void cvtT_f32_f16(const float* __restrict__ in, _Float16* __restrict__ out,
                             int K, int N) {
  int i = blockIdx.x * blockDim.x + threadIdx.x;
  if (i >= K * N) return;
  int k = i / N, n = i % N;
  out[(size_t)n * K + k] = (_Float16)in[i];
}

// ---------------------------------------------------------------------------
// WMMA fragment loaders (wave32; contiguous-K order, same for A and B)
// A 16x32 f16: lane row = lane&15; elems 0..15 -> K = klo..klo+15,
//              klo = (lane>>4)*16 (contiguous along K)
// ---------------------------------------------------------------------------
__device__ __forceinline__ v16h load_a16(const _Float16* A, int lda, int row0, int k0, int lane) {
  const _Float16* p = A + (size_t)(row0 + (lane & 15)) * lda + k0 + ((lane >> 4) << 4);
  v8h lo = *(const v8h*)(p);
  v8h hi = *(const v8h*)(p + 8);
  v16h r;
#pragma unroll
  for (int e = 0; e < 8; ++e) { r[e] = lo[e]; r[8 + e] = hi[e]; }
  return r;
}

// B 32x16 f16 from Bt stored [N][K]: lane col = lane&15;
// elems 0..15 -> K = k0 + (lane>>4)*16 + e  (contiguous along K)
__device__ __forceinline__ v16h load_bT16(const _Float16* Bt, int ldb, int col0, int k0, int lane) {
  const _Float16* p = Bt + (size_t)(col0 + (lane & 15)) * ldb + k0 + ((lane >> 4) << 4);
  v8h lo = *(const v8h*)(p);
  v8h hi = *(const v8h*)(p + 8);
  v16h r;
#pragma unroll
  for (int e = 0; e < 8; ++e) { r[e] = lo[e]; r[8 + e] = hi[e]; }
  return r;
}

// Same B-fragment pattern but from a contiguous 16-half run at p.
__device__ __forceinline__ v16h load_b_run(const _Float16* p) {
  v8h lo = *(const v8h*)(p);
  v8h hi = *(const v8h*)(p + 8);
  v16h r;
#pragma unroll
  for (int e = 0; e < 8; ++e) { r[e] = lo[e]; r[8 + e] = hi[e]; }
  return r;
}

// ---------------------------------------------------------------------------
// WMMA GEMM, 32(M) x 64(N) wave tile: 2 A-frags x 4 B-frags -> 8 WMMAs
// per K-step of 32 (12 b128 loads / 8 WMMAs). C = A[M,K] x Bt[N,K] (+bias).
// ---------------------------------------------------------------------------
__global__ void __launch_bounds__(256) gemm_wmma(
    const _Float16* __restrict__ A, const _Float16* __restrict__ Bt,
    float* __restrict__ Cf, _Float16* __restrict__ Ch,
    const float* __restrict__ bias, int M, int N, int K) {
  int lane = threadIdx.x & 31;
  int wid  = blockIdx.x * (blockDim.x >> 5) + (threadIdx.x >> 5);
  int tn = N >> 6;                          // 64-wide column macro-tiles
  int ntiles = (M >> 5) * tn;               // 32-high row macro-tiles
  if (wid >= ntiles) return;                // whole-wave uniform: EXEC stays all-ones
  int row0 = (wid / tn) << 5;
  int col0 = (wid % tn) << 6;

  v8f acc[2][4];
#pragma unroll
  for (int mi = 0; mi < 2; ++mi)
#pragma unroll
    for (int j = 0; j < 4; ++j) acc[mi][j] = (v8f){};

  for (int k = 0; k < K; k += 32) {
    v16h a0 = load_a16(A, K, row0,      k, lane);
    v16h a1 = load_a16(A, K, row0 + 16, k, lane);
#pragma unroll
    for (int j = 0; j < 4; ++j) {
      v16h b = load_bT16(Bt, K, col0 + 16 * j, k, lane);
      acc[0][j] = __builtin_amdgcn_wmma_f32_16x16x32_f16(false, a0, false, b, (short)0, acc[0][j], false, false);
      acc[1][j] = __builtin_amdgcn_wmma_f32_16x16x32_f16(false, a1, false, b, (short)0, acc[1][j], false, false);
    }
  }

  int collane = lane & 15;
  int rgrp    = (lane >> 4) << 3;
#pragma unroll
  for (int mi = 0; mi < 2; ++mi) {
#pragma unroll
    for (int j = 0; j < 4; ++j) {
      int col = col0 + 16 * j + collane;
      float bv = bias ? bias[col] : 0.0f;
#pragma unroll
      for (int r = 0; r < 8; ++r) {
        size_t idx = (size_t)(row0 + 16 * mi + rgrp + r) * N + col;
        float v = acc[mi][j][r] + bv;
        if (Ch) Ch[idx] = (_Float16)v;
        else    Cf[idx] = v;
      }
    }
  }
}

// ---------------------------------------------------------------------------
// RoPE (interleaved pairs) for Q: qf32[B,S,H,D] -> q16[B,H,S,D]
// ---------------------------------------------------------------------------
__global__ void rope_q_kernel(const float* __restrict__ q, _Float16* __restrict__ q16) {
  const int HP = HEAD_DIM / 2;
  int idx = blockIdx.x * blockDim.x + threadIdx.x;
  if (idx >= B_SZ * S_LEN * N_HEAD * HP) return;
  int i = idx % HP;
  int h = (idx / HP) % N_HEAD;
  int s = (idx / (HP * N_HEAD)) % S_LEN;
  int b = idx / (HP * N_HEAD * S_LEN);
  float inv = __powf(10000.0f, -(2.0f * i) / (float)HEAD_DIM);
  float sn, cs;
  __sincosf((float)s * inv, &sn, &cs);
  size_t src = ((size_t)((b * S_LEN + s) * N_HEAD + h)) * HEAD_DIM + 2 * i;
  float x1 = q[src], x2 = q[src + 1];
  size_t dst = ((size_t)((b * N_HEAD + h) * S_LEN + s)) * HEAD_DIM + 2 * i;
  q16[dst]     = (_Float16)(x1 * cs - x2 * sn);
  q16[dst + 1] = (_Float16)(x1 * sn + x2 * cs);
}

// RoPE K + convert V.
// K: kvf32[B,S,KVH,2D] -> k16[B,KVH,S,D]  (rope, row-major along D)
// V: -> v16T[B,KVH,D,S] (TRANSPOSED so PV B-fragments load contiguously)
__global__ void rope_kv_kernel(const float* __restrict__ kv,
                               _Float16* __restrict__ k16, _Float16* __restrict__ v16T) {
  const int HP = HEAD_DIM / 2;
  int idx = blockIdx.x * blockDim.x + threadIdx.x;
  if (idx >= B_SZ * S_LEN * KV_HEADS * HP) return;
  int i  = idx % HP;
  int kh = (idx / HP) % KV_HEADS;
  int s  = (idx / (HP * KV_HEADS)) % S_LEN;
  int b  = idx / (HP * KV_HEADS * S_LEN);
  float inv = __powf(10000.0f, -(2.0f * i) / (float)HEAD_DIM);
  float sn, cs;
  __sincosf((float)s * inv, &sn, &cs);
  size_t src = ((size_t)((b * S_LEN + s) * KV_HEADS + kh)) * (2 * HEAD_DIM);
  float k1 = kv[src + 2 * i], k2 = kv[src + 2 * i + 1];
  float vv1 = kv[src + HEAD_DIM + 2 * i], vv2 = kv[src + HEAD_DIM + 2 * i + 1];
  size_t kdst = ((size_t)((b * KV_HEADS + kh) * S_LEN + s)) * HEAD_DIM + 2 * i;
  k16[kdst]     = (_Float16)(k1 * cs - k2 * sn);
  k16[kdst + 1] = (_Float16)(k1 * sn + k2 * cs);
  size_t vbase = ((size_t)(b * KV_HEADS + kh) * HEAD_DIM) * S_LEN;
  v16T[vbase + (size_t)(2 * i)     * S_LEN + s] = (_Float16)vv1;
  v16T[vbase + (size_t)(2 * i + 1) * S_LEN + s] = (_Float16)vv2;
}

// ---------------------------------------------------------------------------
// Flash attention (causal, GQA). 4 waves/block; wave owns 16 query rows.
// Per 32-wide K block: the shared 32x128 K tile is staged once into LDS via
// CDNA5 async global->LDS copies (all 4 waves consume it), scores = 8 WMMAs,
// P->LDS->A-fragment, PV = 8 WMMAs with V read from the transposed layout.
// ---------------------------------------------------------------------------
__global__ void __launch_bounds__(128) flash_attn(
    const _Float16* __restrict__ q16, const _Float16* __restrict__ k16,
    const _Float16* __restrict__ v16T, _Float16* __restrict__ o16) {
  __shared__ __align__(32) _Float16 kt[32][HEAD_DIM];   // 8KB shared K tile
  __shared__ __align__(32) _Float16 pls[4][16][32];     // 4KB P staging (per wave)
  int lane = threadIdx.x & 31;
  int wv   = threadIdx.x >> 5;
  int tid  = threadIdx.x;

  int blk = blockIdx.x;                       // B * H * (S/64)
  int qb  = blk % (S_LEN / 64);
  int h   = (blk / (S_LEN / 64)) % N_HEAD;
  int b   = blk / ((S_LEN / 64) * N_HEAD);
  int kvh = h / (N_HEAD / KV_HEADS);
  int qrow0 = qb * 64 + wv * 16;

  const _Float16* Q   = q16  + ((size_t)(b * N_HEAD   + h  ) * S_LEN) * HEAD_DIM;
  const _Float16* Kp  = k16  + ((size_t)(b * KV_HEADS + kvh) * S_LEN) * HEAD_DIM;
  const _Float16* VpT = v16T + ((size_t)(b * KV_HEADS + kvh) * HEAD_DIM) * S_LEN;

  v16h qf[4];
#pragma unroll
  for (int c = 0; c < 4; ++c) qf[c] = load_a16(Q, HEAD_DIM, qrow0, c * 32, lane);

  v8f o[8] = {};
  float m_i[8], l_i[8];
#pragma unroll
  for (int r = 0; r < 8; ++r) { m_i[r] = -INFINITY; l_i[r] = 0.0f; }

  const float scale = 0.08838834764831845f;   // 1/sqrt(128)
  int collane = lane & 15;
  int rgrp    = (lane >> 4) << 3;             // +0 or +8 row group
  int nkb = qb * 2 + 2;                       // uniform per block -> barriers legal

  for (int kb = 0; kb < nkb; ++kb) {
    int kstart = kb * 32;

    // ---- stage K tile (32 rows x 128 halfs = 512 x 16B) into LDS, async
#pragma unroll
    for (int it = 0; it < 4; ++it) {
      int t   = tid + it * 128;               // 0..511
      int row = t >> 4;                       // 16 x b128 per row
      int ch  = (t & 15) << 3;                // half offset within row
      async_cp_b128((unsigned)(size_t)&kt[row][ch],
                    Kp + (size_t)(kstart + row) * HEAD_DIM + ch);
    }
    wait_asynccnt0();
    __syncthreads();

    // ---- scores: two 16-col tiles, K-dim 128 in 4 WMMA steps each (from LDS)
    v8f sc[2] = {};
#pragma unroll
    for (int t = 0; t < 2; ++t) {
#pragma unroll
      for (int c = 0; c < 4; ++c) {
        v16h bk = load_b_run(&kt[t * 16 + collane][c * 32 + ((lane >> 4) << 4)]);
        sc[t] = __builtin_amdgcn_wmma_f32_16x16x32_f16(false, qf[c], false, bk,
                                                       (short)0, sc[t], false, false);
      }
    }

    // ---- scale + causal mask
#pragma unroll
    for (int t = 0; t < 2; ++t) {
      int j = kstart + t * 16 + collane;
#pragma unroll
      for (int r = 0; r < 8; ++r) {
        float vsc = sc[t][r] * scale;
        sc[t][r] = (j > qrow0 + rgrp + r) ? -INFINITY : vsc;
      }
    }

    // ---- online softmax update (row stats replicated across 16-lane groups)
    float p0[8], p1[8];
#pragma unroll
    for (int r = 0; r < 8; ++r) {
      float mr = fmaxf(sc[0][r], sc[1][r]);
      for (int off = 1; off < 16; off <<= 1) mr = fmaxf(mr, __shfl_xor(mr, off, 32));
      float nm = fmaxf(m_i[r], mr);
      float alpha = __expf(m_i[r] - nm);
      float e0 = __expf(sc[0][r] - nm);
      float e1 = __expf(sc[1][r] - nm);
      float rs = e0 + e1;
      for (int off = 1; off < 16; off <<= 1) rs += __shfl_xor(rs, off, 32);
      l_i[r] = l_i[r] * alpha + rs;
      m_i[r] = nm;
      p0[r] = e0; p1[r] = e1;
#pragma unroll
      for (int c = 0; c < 8; ++c) o[c][r] *= alpha;
    }

    // ---- stage P (C-layout) into LDS as f16
#pragma unroll
    for (int r = 0; r < 8; ++r) {
      int row = r + rgrp;
      pls[wv][row][collane]      = (_Float16)p0[r];
      pls[wv][row][16 + collane] = (_Float16)p1[r];
    }
    __syncthreads();

    // ---- reload P in A-fragment layout (16x32)
    v16h pf;
    {
      const _Float16* p = &pls[wv][lane & 15][(lane >> 4) << 4];
      v8h lo = *(const v8h*)(p);
      v8h hi = *(const v8h*)(p + 8);
#pragma unroll
      for (int e = 0; e < 8; ++e) { pf[e] = lo[e]; pf[8 + e] = hi[e]; }
    }
    __syncthreads();

    // ---- O += P @ V  (V transposed: contiguous 16-half runs along keys)
#pragma unroll
    for (int c = 0; c < 8; ++c) {
      v16h bv = load_b_run(VpT + (size_t)(c * 16 + collane) * S_LEN
                               + kstart + ((lane >> 4) << 4));
      o[c] = __builtin_amdgcn_wmma_f32_16x16x32_f16(false, pf, false, bv,
                                                    (short)0, o[c], false, false);
    }
  }

  // ---- finalize: divide by row sums, store [BS, E] f16 (col = h*D + d)
#pragma unroll
  for (int r = 0; r < 8; ++r) {
    float inv_l = 1.0f / l_i[r];
    size_t row = (size_t)b * S_LEN + (qrow0 + rgrp + r);
#pragma unroll
    for (int c = 0; c < 8; ++c) {
      o16[row * N_EMBED + h * HEAD_DIM + c * 16 + collane] = (_Float16)(o[c][r] * inv_l);
    }
  }
}

// ---------------------------------------------------------------------------
// Host launch
// ---------------------------------------------------------------------------
extern "C" void kernel_launch(void* const* d_in, const int* in_sizes, int n_in,
                              void* d_out, int out_size, void* d_ws, size_t ws_size,
                              hipStream_t stream) {
  const float* x     = (const float*)d_in[0];
  const float* q_a_w = (const float*)d_in[1];   // [2048, 256]
  const float* q_b_w = (const float*)d_in[2];   // [256, 2048]
  const float* kv_a_w= (const float*)d_in[3];   // [2048, 128]
  const float* kv_b_w= (const float*)d_in[4];   // [128, 1024]
  const float* out_w = (const float*)d_in[5];   // [2048, 2048]
  const float* out_b = (const float*)d_in[6];   // [2048]
  float* out = (float*)d_out;

  char* ws = (char*)d_ws;
  size_t off = 0;
  auto alloc = [&](size_t bytes) -> char* {
    char* p = ws + off;
    off += (bytes + 255) & ~(size_t)255;
    return p;
  };

  _Float16* x16     = (_Float16*)alloc((size_t)BS_TOK * N_EMBED * 2);
  _Float16* qa_t    = (_Float16*)alloc((size_t)Q_RANK * N_EMBED * 2);
  _Float16* qb_t    = (_Float16*)alloc((size_t)N_EMBED * Q_RANK * 2);
  _Float16* kva_t   = (_Float16*)alloc((size_t)KV_RANK * N_EMBED * 2);
  _Float16* kvb_t   = (_Float16*)alloc((size_t)(KV_HEADS * 2 * HEAD_DIM) * KV_RANK * 2);
  _Float16* ow_t    = (_Float16*)alloc((size_t)N_EMBED * N_EMBED * 2);
  _Float16* qlat16  = (_Float16*)alloc((size_t)BS_TOK * Q_RANK * 2);
  _Float16* kvlat16 = (_Float16*)alloc((size_t)BS_TOK * KV_RANK * 2);
  float*    qf32    = (float*)   alloc((size_t)BS_TOK * N_EMBED * 4);
  float*    kvf32   = (float*)   alloc((size_t)BS_TOK * KV_HEADS * 2 * HEAD_DIM * 4);
  _Float16* q16     = (_Float16*)alloc((size_t)BS_TOK * N_EMBED * 2);
  _Float16* k16     = (_Float16*)alloc((size_t)BS_TOK * KV_HEADS * HEAD_DIM * 2);
  _Float16* v16T    = (_Float16*)alloc((size_t)BS_TOK * KV_HEADS * HEAD_DIM * 2);
  _Float16* attn16  = (_Float16*)qf32;   // alias: qf32 dead after rope_q

  // 1) converts
  {
    int n = BS_TOK * N_EMBED;
    cvt_f32_f16<<<(n + 255) / 256, 256, 0, stream>>>(x, x16, n);
  }
  cvtT_f32_f16<<<(N_EMBED * Q_RANK + 255) / 256, 256, 0, stream>>>(q_a_w, qa_t, N_EMBED, Q_RANK);
  cvtT_f32_f16<<<(Q_RANK * N_EMBED + 255) / 256, 256, 0, stream>>>(q_b_w, qb_t, Q_RANK, N_EMBED);
  cvtT_f32_f16<<<(N_EMBED * KV_RANK + 255) / 256, 256, 0, stream>>>(kv_a_w, kva_t, N_EMBED, KV_RANK);
  cvtT_f32_f16<<<(KV_RANK * 1024 + 255) / 256, 256, 0, stream>>>(kv_b_w, kvb_t, KV_RANK, 1024);
  cvtT_f32_f16<<<(N_EMBED * N_EMBED + 255) / 256, 256, 0, stream>>>(out_w, ow_t, N_EMBED, N_EMBED);

  auto gemm = [&](const _Float16* A, const _Float16* Bt, float* Cf, _Float16* Ch,
                  const float* bias, int M, int N, int K) {
    int ntiles = (M / 32) * (N / 64);
    int blocks = (ntiles + 7) / 8;           // 8 waves (256 threads) per block
    gemm_wmma<<<blocks, 256, 0, stream>>>(A, Bt, Cf, Ch, bias, M, N, K);
  };

  // 2) projections
  gemm(x16,     qa_t,  nullptr, qlat16,  nullptr, BS_TOK, Q_RANK,  N_EMBED);
  gemm(qlat16,  qb_t,  qf32,    nullptr, nullptr, BS_TOK, N_EMBED, Q_RANK);
  gemm(x16,     kva_t, nullptr, kvlat16, nullptr, BS_TOK, KV_RANK, N_EMBED);
  gemm(kvlat16, kvb_t, kvf32,   nullptr, nullptr, BS_TOK, 1024,    KV_RANK);

  // 3) RoPE + layout change (V transposed for contiguous PV fragments)
  {
    int n = B_SZ * S_LEN * N_HEAD * (HEAD_DIM / 2);
    rope_q_kernel<<<(n + 255) / 256, 256, 0, stream>>>(qf32, q16);
  }
  {
    int n = B_SZ * S_LEN * KV_HEADS * (HEAD_DIM / 2);
    rope_kv_kernel<<<(n + 255) / 256, 256, 0, stream>>>(kvf32, k16, v16T);
  }

  // 4) flash attention (writes attn16, aliased over dead qf32)
  {
    int blocks = B_SZ * N_HEAD * (S_LEN / 64);
    flash_attn<<<blocks, 128, 0, stream>>>(q16, k16, v16T, attn16);
  }

  // 5) output projection + bias -> f32 result
  gemm(attn16, ow_t, out, nullptr, out_b, BS_TOK, N_EMBED, N_EMBED);
}
